// Cross_classifier_77627238908199
// MI455X (gfx1250) — compile-verified, matrix operand order checked by reading, not run on hardware
//
#include <hip/hip_runtime.h>
#include <hip/hip_bf16.h>
#include <math.h>

// ---------------------------------------------------------------------------
// CDNA5 / gfx1250 implementation.
//   - wave32, WMMA f32_16x16x32_f16 for both GEMMs (projection + implicit-GEMM conv)
//   - conv computed ONLY for the center 8x8 crop (4x FLOP saving, no padding paths)
//   - x pre-cast to f16 once; conv inner loop is pure b128 loads + 2 WMMA
//   - fused LN+GELU epilogue, fused BN+GELU+xcorr-dot epilogue
// ---------------------------------------------------------------------------

#define EMBED 768
#define E2    384
#define BATCH 128
#define TT    256
#define T1    64
#define EPSLN 1e-5f

typedef __attribute__((ext_vector_type(16))) _Float16 v16h;
typedef __attribute__((ext_vector_type(8)))  float    v8f;

union H2   { unsigned u; _Float16 h[2]; };
union Frag { v16h v; unsigned u[8]; };

__device__ __forceinline__ unsigned pack2(float a, float b) {
    H2 t; t.h[0] = (_Float16)a; t.h[1] = (_Float16)b; return t.u;
}
__device__ __forceinline__ float gelu_exact(float x) {
    return 0.5f * x * (1.0f + erff(x * 0.70710678118654752f));
}
// A/B fragment (16x32 f16) lane/K packing per CDNA5 ISA 7.12.2:
//   lane L (row = L&15, hi = L>>4) holds K pairs at K = 2*(j&3) + 16*(j>>2) + 8*hi
//   -> in u32 (2-half) units: (j&3) + 8*(j>>2) + 4*hi  (vectorizes to 2x b128)
__device__ __forceinline__ int pair_idx(int j, int hi) {
    return (j & 3) + ((j >> 2) << 3) + (hi << 2);
}

// ---------------------------------------------------------------------------
// prep kernels
// ---------------------------------------------------------------------------
__global__ void cast_f16_kernel(const float* __restrict__ in, _Float16* __restrict__ out, int n) {
    int i = blockIdx.x * blockDim.x + threadIdx.x;
    if (i < n) out[i] = (_Float16)in[i];
}

// bulk fp32 -> f16 cast, 4 elements/thread (float4 in, uint2 out)
__global__ void cast_x_kernel(const float4* __restrict__ in, uint2* __restrict__ out, int n4) {
    int i = blockIdx.x * blockDim.x + threadIdx.x;
    if (i < n4) {
        float4 v = in[i];
        uint2 r;
        r.x = pack2(v.x, v.y);
        r.y = pack2(v.z, v.w);
        out[i] = r;
    }
}

// w [384][768][3][3] (o,i,tap) -> out [9][384][768] (tap,o,i) so K (=i) is contiguous per o-row
__global__ void transpose_w_kernel(const float* __restrict__ w, _Float16* __restrict__ out) {
    int i = blockIdx.x * blockDim.x + threadIdx.x;
    const int total = E2 * EMBED * 9;
    if (i >= total) return;
    int tap = i % 9;
    int rest = i / 9;
    int ic = rest % EMBED;
    int o  = rest / EMBED;
    out[((size_t)tap * E2 + o) * EMBED + ic] = (_Float16)w[i];
}

__global__ void zero_acc_kernel(float* __restrict__ acc) { acc[threadIdx.x] = 0.0f; }

__global__ void finalize_kernel(const float* __restrict__ acc, const float* __restrict__ c,
                                float* __restrict__ out) {
    int i = threadIdx.x;              // 0..255 : [0,128) = s1, [128,256) = s2
    float v = acc[i] / c[0];
    out[i] = 1.0f / (1.0f + expf(-v));
}

// ---------------------------------------------------------------------------
// GEMM1: z = GELU(LN(concat(z_r,z_i) @ fz_w^T + fz_b))  ->  z_f[b][o][t]  (fp32)
// block = 256 thr (8 waves); block owns 16 rows x 384 cols; wave w owns cols [48w,48w+48)
// ---------------------------------------------------------------------------
__global__ void __launch_bounds__(256)
gemm_z_kernel(const float* __restrict__ z_r, const float* __restrict__ z_i,
              const unsigned* __restrict__ fzw_h,     // [384][768] u32 (=1536 f16 per row)
              const float* __restrict__ fz_b, const float* __restrict__ ln_g,
              const float* __restrict__ ln_b, float* __restrict__ z_f) {
    __shared__ unsigned lds_a[16][16];     // 16 rows x 32 halves (u32-packed)
    __shared__ float    ybuf[16][E2];
    __shared__ float    red_s[16][16], red_q[16][16];
    __shared__ float    mean_s[16], rstd_s[16];

    const int tid  = threadIdx.x;
    const int mb   = blockIdx.x;          // 0..511
    const int b    = mb >> 2;             // 4 row-blocks of 16 per batch (64 rows)
    const int t0   = (mb & 3) << 4;
    const int wave = tid >> 5, lane = tid & 31;
    const int lr = lane & 15, hi = lane >> 4;

    v8f acc[3] = {};

    const int arow  = tid >> 4;           // staging: one u32 (2 halves) per thread
    const int apair = tid & 15;

    #pragma unroll 1
    for (int kb = 0; kb < 48; ++kb) {
        const int k0 = kb * 32;
        // ---- stage A tile (fp32 -> f16) into LDS ----
        const int e = k0 + apair * 2;                 // even; never straddles 768
        const int t = t0 + arow;
        float f0, f1;
        if (e < EMBED) {
            const float* p = &z_r[((size_t)b * T1 + t) * EMBED + e];
            f0 = p[0]; f1 = p[1];
        } else {
            const float* p = &z_i[((size_t)b * T1 + t) * EMBED + (e - EMBED)];
            f0 = p[0]; f1 = p[1];
        }
        lds_a[arow][apair] = pack2(f0, f1);
        __syncthreads();

        Frag a;
        #pragma unroll
        for (int j = 0; j < 8; ++j) a.u[j] = lds_a[lr][pair_idx(j, hi)];

        #pragma unroll
        for (int nt = 0; nt < 3; ++nt) {
            const int o = wave * 48 + nt * 16 + lr;
            const unsigned* browp = &fzw_h[(size_t)o * (EMBED) + (k0 >> 1)]; // 768 u32 per row
            Frag bf;
            #pragma unroll
            for (int j = 0; j < 8; ++j) bf.u[j] = browp[pair_idx(j, hi)];
            acc[nt] = __builtin_amdgcn_wmma_f32_16x16x32_f16(
                false, a.v, false, bf.v, (short)0, acc[nt], false, false);
        }
        __syncthreads();
    }

    // ---- spill D + bias to LDS (D layout: VGPR r -> M=r+8*hi, N=lane&15) ----
    #pragma unroll
    for (int nt = 0; nt < 3; ++nt) {
        #pragma unroll
        for (int r = 0; r < 8; ++r) {
            const int mrow = r + (hi << 3);
            const int ncol = wave * 48 + nt * 16 + lr;
            ybuf[mrow][ncol] = acc[nt][r] + fz_b[ncol];
        }
    }
    __syncthreads();

    // ---- LayerNorm over 384 per row + GELU ----
    const int row = tid >> 4, seg = tid & 15;        // 16 threads per row, 24 cols each
    float s = 0.f, q = 0.f;
    #pragma unroll 1
    for (int j = 0; j < 24; ++j) { float v = ybuf[row][seg * 24 + j]; s += v; q += v * v; }
    red_s[row][seg] = s; red_q[row][seg] = q;
    __syncthreads();
    if (seg == 0) {
        float ts = 0.f, tq = 0.f;
        #pragma unroll
        for (int j = 0; j < 16; ++j) { ts += red_s[row][j]; tq += red_q[row][j]; }
        const float mu  = ts * (1.0f / E2);
        const float var = tq * (1.0f / E2) - mu * mu;
        mean_s[row] = mu;
        rstd_s[row] = rsqrtf(var + EPSLN);
    }
    __syncthreads();
    const float mu = mean_s[row], rs = rstd_s[row];
    const int   t  = t0 + row;
    #pragma unroll 1
    for (int j = 0; j < 24; ++j) {
        const int n = seg * 24 + j;
        float v = (ybuf[row][n] - mu) * rs * ln_g[n] + ln_b[n];
        v = gelu_exact(v);
        z_f[((size_t)b * E2 + n) * T1 + t] = v;      // layout [b][o][t]
    }
}

// ---------------------------------------------------------------------------
// Implicit-GEMM 3x3 conv (center 8x8 only) + BN + GELU + xcorr dot, per batch.
// grid = (12 o-tile-pairs, 128 batches), block = 128 thr (4 waves).
// wave = one 16-position M-tile x 32 out-channels (2 WMMAs per K-step, A reused).
// K = 9 taps x 768 channels; all operands pre-cast f16 -> inner loop is 6x b128 + 2 wmma.
// ---------------------------------------------------------------------------
__global__ void __launch_bounds__(128)
conv_xcorr_kernel(const unsigned* __restrict__ xh,       // [128][256][384] u32 (=768 f16/row)
                  const unsigned* __restrict__ wh,       // [9][384][384] u32 tap-major
                  const float* __restrict__ bconv, const float* __restrict__ bn_g,
                  const float* __restrict__ bn_b, const float* __restrict__ bn_m,
                  const float* __restrict__ bn_v, const float* __restrict__ z_f,
                  float* __restrict__ acc) {
    const int tid   = threadIdx.x;
    const int ntile = blockIdx.x;                     // 0..11 -> out channels [32*nt, 32*nt+32)
    const int b     = blockIdx.y;
    const int wave  = tid >> 5, lane = tid & 31, lr = lane & 15, hi = lane >> 4;

    const int p  = wave * 16 + lr;                    // A-row position in center 8x8 (0..63)
    const int ph = (p >> 3) + 4, pw = (p & 7) + 4;
    const int o0 = ntile * 32 + lr;                   // B-row out channels for this lane
    const int o1 = o0 + 16;

    v8f c0 = {}, c1 = {};

    #pragma unroll 1
    for (int tap = 0; tap < 9; ++tap) {
        const int kh = tap / 3 - 1, kw = tap % 3 - 1;
        const int sp = (ph + kh) * 16 + (pw + kw);    // always interior (3..12)
        const unsigned* xrow = &xh[((size_t)b * TT + sp) * (EMBED / 2)];
        const unsigned* w0   = &wh[((size_t)tap * E2 + o0) * (EMBED / 2)];
        const unsigned* w1   = &wh[((size_t)tap * E2 + o1) * (EMBED / 2)];
        #pragma unroll 2
        for (int cb = 0; cb < 24; ++cb) {
            const int k2 = cb * 16;                   // K offset in u32 units
            __builtin_prefetch(xrow + k2 + 32, 0, 0); // global_prefetch of next chunk
            Frag a, bf0, bf1;
            #pragma unroll
            for (int j = 0; j < 8; ++j) {
                const int idx = k2 + pair_idx(j, hi);
                a.u[j]   = xrow[idx];
                bf0.u[j] = w0[idx];
                bf1.u[j] = w1[idx];
            }
            c0 = __builtin_amdgcn_wmma_f32_16x16x32_f16(
                false, a.v, false, bf0.v, (short)0, c0, false, false);
            c1 = __builtin_amdgcn_wmma_f32_16x16x32_f16(
                false, a.v, false, bf1.v, (short)0, c1, false, false);
        }
    }

    // ---- fused epilogue: +bias, BN, GELU, dot with z_f (both N-tiles) ----
    float partial = 0.f;
    #pragma unroll
    for (int half = 0; half < 2; ++half) {
        const int   o     = half ? o1 : o0;
        const v8f   c8    = half ? c1 : c0;
        const float bb    = bconv[o];
        const float rsbn  = rsqrtf(bn_v[o] + EPSLN);
        const float scale = rsbn * bn_g[o];
        const float beta  = bn_b[o];
        const float mean  = bn_m[o];
        #pragma unroll
        for (int r = 0; r < 8; ++r) {
            const int pl = wave * 16 + r + (hi << 3); // D: M = r + 8*hi
            float yv = c8[r] + bb;
            yv = (yv - mean) * scale + beta;
            yv = gelu_exact(yv);
            partial += yv * z_f[((size_t)b * E2 + o) * T1 + pl];
        }
    }

    __shared__ float red[128];
    red[tid] = partial;
    __syncthreads();
    #pragma unroll
    for (int s2 = 64; s2 > 0; s2 >>= 1) {
        if (tid < s2) red[tid] += red[tid + s2];
        __syncthreads();
    }
    if (tid == 0) atomicAdd(&acc[b], red[0]);
}

// ---------------------------------------------------------------------------
// host launcher
// ---------------------------------------------------------------------------
extern "C" void kernel_launch(void* const* d_in, const int* in_sizes, int n_in,
                              void* d_out, int out_size, void* d_ws, size_t ws_size,
                              hipStream_t stream) {
    const float* z_r   = (const float*)d_in[0];
    const float* z_i   = (const float*)d_in[1];
    const float* x_r   = (const float*)d_in[2];
    const float* x_i   = (const float*)d_in[3];
    const float* fz_w  = (const float*)d_in[4];
    const float* fz_b  = (const float*)d_in[5];
    const float* ln_g  = (const float*)d_in[6];
    const float* ln_b  = (const float*)d_in[7];
    const float* wr    = (const float*)d_in[8];
    const float* br    = (const float*)d_in[9];
    const float* bnr_g = (const float*)d_in[10];
    const float* bnr_b = (const float*)d_in[11];
    const float* bnr_m = (const float*)d_in[12];
    const float* bnr_v = (const float*)d_in[13];
    const float* wi    = (const float*)d_in[14];
    const float* bi    = (const float*)d_in[15];
    const float* bni_g = (const float*)d_in[16];
    const float* bni_b = (const float*)d_in[17];
    const float* bni_m = (const float*)d_in[18];
    const float* bni_v = (const float*)d_in[19];
    const float* c     = (const float*)d_in[20];

    // workspace carve-up (~125 MB)
    char*  ws    = (char*)d_ws;
    size_t off   = 0;
    _Float16* fzw_h = (_Float16*)(ws + off); off += (size_t)E2 * 2 * EMBED * sizeof(_Float16);
    _Float16* wh_r  = (_Float16*)(ws + off); off += (size_t)9 * E2 * EMBED * sizeof(_Float16);
    _Float16* wh_i  = (_Float16*)(ws + off); off += (size_t)9 * E2 * EMBED * sizeof(_Float16);
    _Float16* xh_r  = (_Float16*)(ws + off); off += (size_t)BATCH * TT * EMBED * sizeof(_Float16);
    _Float16* xh_i  = (_Float16*)(ws + off); off += (size_t)BATCH * TT * EMBED * sizeof(_Float16);
    float*    z_f   = (float*)(ws + off);    off += (size_t)BATCH * E2 * T1 * sizeof(float);
    float*    acc   = (float*)(ws + off);    off += 256 * sizeof(float);

    zero_acc_kernel<<<1, 256, 0, stream>>>(acc);

    const int n_fzw = E2 * 2 * EMBED;
    cast_f16_kernel<<<(n_fzw + 255) / 256, 256, 0, stream>>>(fz_w, fzw_h, n_fzw);
    const int n_w = E2 * EMBED * 9;
    transpose_w_kernel<<<(n_w + 255) / 256, 256, 0, stream>>>(wr, wh_r);
    transpose_w_kernel<<<(n_w + 255) / 256, 256, 0, stream>>>(wi, wh_i);
    const int n_x4 = (BATCH * TT * EMBED) / 4;
    cast_x_kernel<<<(n_x4 + 255) / 256, 256, 0, stream>>>(
        (const float4*)x_r, (uint2*)xh_r, n_x4);
    cast_x_kernel<<<(n_x4 + 255) / 256, 256, 0, stream>>>(
        (const float4*)x_i, (uint2*)xh_i, n_x4);

    gemm_z_kernel<<<(BATCH * T1) / 16, 256, 0, stream>>>(
        z_r, z_i, (const unsigned*)fzw_h, fz_b, ln_g, ln_b, z_f);

    dim3 cgrid(E2 / 32, BATCH);
    conv_xcorr_kernel<<<cgrid, 128, 0, stream>>>(
        (const unsigned*)xh_r, (const unsigned*)wh_r, br, bnr_g, bnr_b, bnr_m, bnr_v, z_f, acc);
    conv_xcorr_kernel<<<cgrid, 128, 0, stream>>>(
        (const unsigned*)xh_i, (const unsigned*)wh_i, bi, bni_g, bni_b, bni_m, bni_v, z_f, acc + 128);

    finalize_kernel<<<1, 256, 0, stream>>>(acc, c, (float*)d_out);
}
